// MultiHeadAttentionFromScratch_25039659336135
// MI455X (gfx1250) — matrix operand order repaired and checked by reading, not compiled, not run on hardware
//
#include <hip/hip_runtime.h>
#include <hip/hip_bf16.h>

#define B_ 8
#define S_ 1024
#define E_ 1024
#define H_ 16
#define D_ 64

typedef __attribute__((ext_vector_type(16))) __bf16 v16bf;
typedef __attribute__((ext_vector_type(8)))  __bf16 v8bf;
typedef __attribute__((ext_vector_type(8)))  float  v8f;
typedef __attribute__((ext_vector_type(4)))  float  v4f;
typedef __attribute__((ext_vector_type(4)))  int    v4i;

// ---------------------------------------------------------------------------
// Async global->LDS path (gfx1250 GLOBAL_LOAD_ASYNC_TO_LDS_B128, ASYNCcnt).
// Guarded: falls back to plain LDS staging if the builtins are absent.
#if defined(__has_builtin)
#if __has_builtin(__builtin_amdgcn_global_load_async_to_lds_b128) && \
    __has_builtin(__builtin_amdgcn_s_wait_asynccnt)
#define USE_ASYNC_LDS 1
#endif
#endif
#ifndef USE_ASYNC_LDS
#define USE_ASYNC_LDS 0
#endif

#define AS1 __attribute__((address_space(1)))
#define AS3 __attribute__((address_space(3)))

// Copy 16 bytes (8 bf16) global -> LDS (async DMA when available).
static __device__ __forceinline__ void copy16_g2l(__bf16* l, const __bf16* g) {
#if USE_ASYNC_LDS
    // builtin signature: (int4 AS1* gsrc, int4 AS3* ldst, imm offset, imm cpol)
    __builtin_amdgcn_global_load_async_to_lds_b128((AS1 v4i*)(void*)g,
                                                   (AS3 v4i*)l, 0, 0);
#else
    *(v8bf*)l = *(const v8bf*)g;
#endif
}

// Wait for this wave's async copies, then barrier (all waves' copies visible).
static __device__ __forceinline__ void stage_fence() {
#if USE_ASYNC_LDS
    __builtin_amdgcn_s_wait_asynccnt(0);
#endif
    __syncthreads();
}

// ---------------------------------------------------------------------------
// Fragment loaders for V_WMMA_F32_16X16X32_BF16 (wave32), per ISA 7.12.2.
// A operand (16x32 MxK): lane L: m = m0+(L&15), h = L>>4;
//   elems[0..7]=K k0+8h+0..7, elems[8..15]=K k0+16+8h+0..7 (contiguous chunks).
static __device__ __forceinline__
v16bf load_frag_a_bf16(const __bf16* __restrict__ A, int lda, int m0, int k0, int lane) {
    int m = m0 + (lane & 15);
    int h = lane >> 4;
    const __bf16* p0 = A + (size_t)m * lda + k0 + 8 * h;
    v8bf lo = *(const v8bf*)p0;
    v8bf hi = *(const v8bf*)(p0 + 16);
    v16bf r;
#pragma unroll
    for (int i = 0; i < 8; ++i) { r[i] = lo[i]; r[i + 8] = hi[i]; }
    return r;
}

// B operand (32x16 KxN) with B[k][n] = R[n][k] (computing X @ R^T):
//   lane L: n = n0+(L&15), h = L>>4; elems[0..7]=K k0+8h+0..7, elems[8..15]=K k0+16+8h+0..7.
static __device__ __forceinline__
v16bf load_frag_bT_bf16(const __bf16* __restrict__ R, int ldr, int n0, int k0, int lane) {
    int n = n0 + (lane & 15);
    int h = lane >> 4;
    const __bf16* p = R + (size_t)n * ldr + k0 + 8 * h;
    v8bf lo = *(const v8bf*)p;
    v8bf hi = *(const v8bf*)(p + 16);
    v16bf r;
#pragma unroll
    for (int i = 0; i < 8; ++i) { r[i] = lo[i]; r[i + 8] = hi[i]; }
    return r;
}

static __device__ __forceinline__ v8f wmma_bf16(v16bf a, v16bf b, v8f c) {
    return __builtin_amdgcn_wmma_f32_16x16x32_bf16(false, a, false, b,
                                                   (short)0, c, false, false);
}

static __device__ __forceinline__ void store_out(__bf16* p, float v) { *p = (__bf16)v; }
static __device__ __forceinline__ void store_out(float*  p, float v) { *p = v; }

// ---------------------------------------------------------------------------
__global__ __launch_bounds__(256) void cast_f32_bf16(const float* __restrict__ x,
                                                     __bf16* __restrict__ y, int n) {
    int i = blockIdx.x * 256 + threadIdx.x;
    if (i < n) y[i] = (__bf16)x[i];
}

// ---------------------------------------------------------------------------
// Stage one 128x32 bf16 tile of X and one of W into LDS (512 chunks each).
static __device__ __forceinline__
void stage_gemm_tiles(__bf16* Ad, __bf16* Bd,
                      const __bf16* __restrict__ X, const __bf16* __restrict__ W,
                      int m_base, int n_base, int K, int k0, int tid) {
#pragma unroll
    for (int c = tid; c < 512; c += 256) {
        int r = c >> 2, sub = (c & 3) << 3;   // 4 chunks per 32-elem row
        copy16_g2l(&Ad[r * 32 + sub], X + (size_t)(m_base + r) * K + k0 + sub);
        copy16_g2l(&Bd[r * 32 + sub], W + (size_t)(n_base + r) * K + k0 + sub);
    }
}

// Y[M,N] = X[M,K] @ W[N,K]^T + bias. 128x128 block tile, BK=32, 8 waves (2x4),
// each wave owns a 64x32 sub-block = 4x2 WMMA tiles. Double-buffered LDS:
// the async DMA of tile k+1 overlaps the 8 WMMAs on tile k.
template <typename OutT>
__global__ __launch_bounds__(256) void gemm_xwT_bias(const __bf16* __restrict__ X,
                                                     const __bf16* __restrict__ W,
                                                     const float* __restrict__ bias,
                                                     OutT* __restrict__ Y,
                                                     int M, int N, int K) {
    __shared__ __bf16 As[2][128 * 32];   // [m][k], ld = 32
    __shared__ __bf16 Bs[2][128 * 32];   // [n][k], ld = 32

    const int tid  = threadIdx.x;
    const int lane = tid & 31;
    const int wave = tid >> 5;
    const int wm = wave >> 2;            // 0..1 -> 64 rows each
    const int wn = wave & 3;             // 0..3 -> 32 cols each

    const int nBlocks = N >> 7;
    const int bm = (int)blockIdx.x / nBlocks;
    const int bn = (int)blockIdx.x - bm * nBlocks;
    const int m_base = bm << 7;
    const int n_base = bn << 7;

    v8f acc[4][2] = {};

    stage_gemm_tiles(As[0], Bs[0], X, W, m_base, n_base, K, 0, tid);
    stage_fence();

    int cur = 0;
    for (int k0 = 0; k0 < K; k0 += 32) {
        int nxt = cur ^ 1;
        if (k0 + 32 < K)   // prefetch next tile; DMA overlaps the WMMAs below
            stage_gemm_tiles(As[nxt], Bs[nxt], X, W, m_base, n_base, K, k0 + 32, tid);

        v16bf a[4], b[2];
#pragma unroll
        for (int mt = 0; mt < 4; ++mt)
            a[mt] = load_frag_a_bf16(As[cur], 32, wm * 64 + mt * 16, 0, lane);
#pragma unroll
        for (int nt = 0; nt < 2; ++nt)
            b[nt] = load_frag_bT_bf16(Bs[cur], 32, wn * 32 + nt * 16, 0, lane);
#pragma unroll
        for (int mt = 0; mt < 4; ++mt)
#pragma unroll
            for (int nt = 0; nt < 2; ++nt)
                acc[mt][nt] = wmma_bf16(a[mt], b[nt], acc[mt][nt]);

        stage_fence();   // drain async + ds reads, release buffers
        cur = nxt;
    }

    const int hh = lane >> 4;
#pragma unroll
    for (int nt = 0; nt < 2; ++nt) {
        int n = n_base + wn * 32 + nt * 16 + (lane & 15);
        float bv = bias[n];
#pragma unroll
        for (int mt = 0; mt < 4; ++mt) {
            int m0 = m_base + wm * 64 + mt * 16;
#pragma unroll
            for (int v = 0; v < 8; ++v)
                store_out(&Y[(size_t)(m0 + v + 8 * hh) * N + n], acc[mt][nt][v] + bv);
        }
    }
}

// ---------------------------------------------------------------------------
// scores[bh,i,j] = (1/8) * sum_d Q[b,i,hD+d] * K[b,j,hD+d]
// 128x128 block tile per head; full K=64 staged once (2 x 16KB LDS).
__global__ __launch_bounds__(256) void attn_scores(const __bf16* __restrict__ Qb,
                                                   const __bf16* __restrict__ Kb,
                                                   float* __restrict__ attn) {
    __shared__ __bf16 Qs[128 * 64];   // [i][d], ld = 64
    __shared__ __bf16 Ks[128 * 64];   // [j][d], ld = 64

    const int tid  = threadIdx.x;
    const int lane = tid & 31;
    const int wave = tid >> 5;
    const int wm = wave >> 2;
    const int wn = wave & 3;

    const int bh = blockIdx.y;
    const int b = bh >> 4, h = bh & 15;
    const int bm = (int)blockIdx.x >> 3;
    const int bn = (int)blockIdx.x & 7;
    const int m_base = bm << 7;
    const int n_base = bn << 7;

    const __bf16* Qh = Qb + (size_t)b * S_ * E_ + h * D_;
    const __bf16* Kh = Kb + (size_t)b * S_ * E_ + h * D_;

    // 128x64 bf16 per matrix = 1024 16B chunks, 4 per thread per matrix.
#pragma unroll
    for (int c = tid; c < 1024; c += 256) {
        int r = c >> 3, sub = (c & 7) << 3;   // 8 chunks per 64-elem row
        copy16_g2l(&Qs[r * 64 + sub], Qh + (size_t)(m_base + r) * E_ + sub);
        copy16_g2l(&Ks[r * 64 + sub], Kh + (size_t)(n_base + r) * E_ + sub);
    }
    stage_fence();

    v8f acc[4][2] = {};
#pragma unroll
    for (int kk = 0; kk < 64; kk += 32) {
        v16bf a[4], b2[2];
#pragma unroll
        for (int mt = 0; mt < 4; ++mt)
            a[mt] = load_frag_a_bf16(Qs, 64, wm * 64 + mt * 16, kk, lane);
#pragma unroll
        for (int nt = 0; nt < 2; ++nt)
            b2[nt] = load_frag_bT_bf16(Ks, 64, wn * 32 + nt * 16, kk, lane);
#pragma unroll
        for (int mt = 0; mt < 4; ++mt)
#pragma unroll
            for (int nt = 0; nt < 2; ++nt)
                acc[mt][nt] = wmma_bf16(a[mt], b2[nt], acc[mt][nt]);
    }

    float* out = attn + (size_t)bh * S_ * S_;
    const int hh = lane >> 4;
#pragma unroll
    for (int mt = 0; mt < 4; ++mt) {
        int m0 = m_base + wm * 64 + mt * 16;
#pragma unroll
        for (int nt = 0; nt < 2; ++nt) {
            int n = n_base + wn * 32 + nt * 16 + (lane & 15);
#pragma unroll
            for (int v = 0; v < 8; ++v)
                out[(size_t)(m0 + v + 8 * hh) * S_ + n] = acc[mt][nt][v] * 0.125f;
        }
    }
}

// ---------------------------------------------------------------------------
// In-place softmax over rows of length S. One wave per row, 8 rows per block.
__global__ __launch_bounds__(256) void softmax_rows(float* __restrict__ attn) {
    int lane = threadIdx.x & 31;
    int wave = threadIdx.x >> 5;
    size_t row = (size_t)blockIdx.x * 8 + wave;
    float* p = attn + row * S_;

    float mx = -3.402823466e+38f;
    for (int j = lane; j < S_; j += 32) mx = fmaxf(mx, p[j]);
#pragma unroll
    for (int o = 16; o > 0; o >>= 1) mx = fmaxf(mx, __shfl_xor(mx, o, 32));

    float sum = 0.f;
    for (int j = lane; j < S_; j += 32) {
        float e = __expf(p[j] - mx);
        p[j] = e;
        sum += e;
    }
#pragma unroll
    for (int o = 16; o > 0; o >>= 1) sum += __shfl_xor(sum, o, 32);

    float inv = 1.0f / sum;
    for (int j = lane; j < S_; j += 32) p[j] *= inv;
}

// ---------------------------------------------------------------------------
// Vt[b,h,d,j] = V[b,j,h*D+d]
__global__ __launch_bounds__(256) void transpose_v(const __bf16* __restrict__ Vb,
                                                   __bf16* __restrict__ Vt) {
    size_t idx = (size_t)blockIdx.x * 256 + threadIdx.x;   // over B*H*D*S
    int j = idx & (S_ - 1);
    int d = (idx >> 10) & (D_ - 1);
    int h = (idx >> 16) & (H_ - 1);
    int b = (int)(idx >> 20);
    Vt[idx] = Vb[((size_t)b * S_ + j) * E_ + h * D_ + d];
}

// ---------------------------------------------------------------------------
// Stage attn (fp32->bf16 convert) and Vt tiles for one K-step of attn@V.
static __device__ __forceinline__
void stage_ctx_tiles(__bf16* Ad, __bf16* Vd,
                     const float* __restrict__ Ah, const __bf16* __restrict__ Bh,
                     int m_base, int k0, int tid) {
    // attn tile: 128x32 fp32 -> bf16, 2 chunks per thread
#pragma unroll
    for (int c = tid; c < 512; c += 256) {
        int r = c >> 2, sub = (c & 3) << 3;
        const float* src = Ah + (size_t)(m_base + r) * S_ + k0 + sub;
        v4f x0 = *(const v4f*)src;
        v4f x1 = *(const v4f*)(src + 4);
        v8bf d;
#pragma unroll
        for (int i = 0; i < 4; ++i) { d[i] = (__bf16)x0[i]; d[i + 4] = (__bf16)x1[i]; }
        *(v8bf*)&Ad[r * 32 + sub] = d;
    }
    // Vt tile: 64x32 bf16 = 256 chunks, exactly 1 per thread (async DMA).
    {
        int r = tid >> 2, sub = (tid & 3) << 3;
        copy16_g2l(&Vd[r * 32 + sub], Bh + (size_t)r * S_ + k0 + sub);
    }
}

// ctx[b,i,h*D+d] = sum_j attn[bh,i,j] * V[b,j,h*D+d]   (V given as Vt[bh,d,j])
// Block tile 128(M) x 64(N=D); waves 4x2, each wave 2x2 WMMA tiles.
// Double-buffered: next tile's staging overlaps current tile's WMMAs.
__global__ __launch_bounds__(256) void attn_ctx(const float* __restrict__ attn,
                                                const __bf16* __restrict__ Vt,
                                                __bf16* __restrict__ ctx) {
    __shared__ __bf16 As[2][128 * 32];   // attn tile, bf16, [i][j], ld = 32
    __shared__ __bf16 Vs[2][64 * 32];    // Vt tile,        [d][j], ld = 32

    const int tid  = threadIdx.x;
    const int lane = tid & 31;
    const int wave = tid >> 5;
    const int wm = wave >> 1;            // 0..3 -> 32 rows each
    const int wn = wave & 1;             // 0..1 -> 32 cols each

    const int bh = blockIdx.y;
    const int m_base = (int)blockIdx.x << 7;

    const float*  Ah = attn + (size_t)bh * S_ * S_;
    const __bf16* Bh = Vt + (size_t)bh * D_ * S_;

    v8f acc[2][2] = {};

    stage_ctx_tiles(As[0], Vs[0], Ah, Bh, m_base, 0, tid);
    stage_fence();

    int cur = 0;
    for (int k0 = 0; k0 < S_; k0 += 32) {
        int nxt = cur ^ 1;
        if (k0 + 32 < S_)
            stage_ctx_tiles(As[nxt], Vs[nxt], Ah, Bh, m_base, k0 + 32, tid);

        v16bf a[2], b[2];
#pragma unroll
        for (int mt = 0; mt < 2; ++mt)
            a[mt] = load_frag_a_bf16(As[cur], 32, wm * 32 + mt * 16, 0, lane);
#pragma unroll
        for (int nt = 0; nt < 2; ++nt)
            b[nt] = load_frag_bT_bf16(Vs[cur], 32, wn * 32 + nt * 16, 0, lane);
#pragma unroll
        for (int mt = 0; mt < 2; ++mt)
#pragma unroll
            for (int nt = 0; nt < 2; ++nt)
                acc[mt][nt] = wmma_bf16(a[mt], b[nt], acc[mt][nt]);

        stage_fence();
        cur = nxt;
    }

    const int b = bh >> 4, h = bh & 15;
    __bf16* out = ctx + (size_t)b * S_ * E_ + h * D_;
    const int hh = lane >> 4;
#pragma unroll
    for (int mt = 0; mt < 2; ++mt) {
        int m0 = m_base + wm * 32 + mt * 16;
#pragma unroll
        for (int nt = 0; nt < 2; ++nt) {
            int n = wn * 32 + nt * 16 + (lane & 15);
#pragma unroll
            for (int v = 0; v < 8; ++v)
                out[(size_t)(m0 + v + 8 * hh) * E_ + n] = (__bf16)acc[mt][nt][v];
        }
    }
}

// ---------------------------------------------------------------------------
extern "C" void kernel_launch(void* const* d_in, const int* in_sizes, int n_in,
                              void* d_out, int out_size, void* d_ws, size_t ws_size,
                              hipStream_t stream) {
    const float* q    = (const float*)d_in[0];
    const float* Wq_w = (const float*)d_in[1];
    const float* Wq_b = (const float*)d_in[2];
    const float* Wk_w = (const float*)d_in[3];
    const float* Wk_b = (const float*)d_in[4];
    const float* Wv_w = (const float*)d_in[5];
    const float* Wv_b = (const float*)d_in[6];
    const float* Wo_w = (const float*)d_in[7];
    const float* Wo_b = (const float*)d_in[8];

    const size_t BSE = (size_t)B_ * S_ * E_;   // 8,388,608
    const size_t EE  = (size_t)E_ * E_;        // 1,048,576

    char* ws = (char*)d_ws;
    size_t off = 0;
    __bf16* qb  = (__bf16*)(ws + off); off += BSE * 2;   // reused as ctx later
    __bf16* wqb = (__bf16*)(ws + off); off += EE * 2;
    __bf16* wkb = (__bf16*)(ws + off); off += EE * 2;
    __bf16* wvb = (__bf16*)(ws + off); off += EE * 2;
    __bf16* wob = (__bf16*)(ws + off); off += EE * 2;
    __bf16* Qb  = (__bf16*)(ws + off); off += BSE * 2;
    __bf16* Kb  = (__bf16*)(ws + off); off += BSE * 2;
    __bf16* Vb  = (__bf16*)(ws + off); off += BSE * 2;
    __bf16* Vt  = (__bf16*)(ws + off); off += BSE * 2;
    __bf16* ctx = qb;  // qb dead after the QKV projections

    float* out_f  = (float*)d_out;              // [B,S,E]
    float* attn_f = out_f + BSE;                // [B,H,S,S]

    // 1) casts
    cast_f32_bf16<<<(int)(BSE / 256), 256, 0, stream>>>(q, qb, (int)BSE);
    cast_f32_bf16<<<(int)(EE / 256), 256, 0, stream>>>(Wq_w, wqb, (int)EE);
    cast_f32_bf16<<<(int)(EE / 256), 256, 0, stream>>>(Wk_w, wkb, (int)EE);
    cast_f32_bf16<<<(int)(EE / 256), 256, 0, stream>>>(Wv_w, wvb, (int)EE);
    cast_f32_bf16<<<(int)(EE / 256), 256, 0, stream>>>(Wo_w, wob, (int)EE);

    // 2) Q/K/V projections: (8192/128)*(1024/128) = 512 blocks
    const int M = B_ * S_;
    const int gemm_blocks = (M / 128) * (E_ / 128);
    gemm_xwT_bias<__bf16><<<gemm_blocks, 256, 0, stream>>>(qb, wqb, Wq_b, Qb, M, E_, E_);
    gemm_xwT_bias<__bf16><<<gemm_blocks, 256, 0, stream>>>(qb, wkb, Wk_b, Kb, M, E_, E_);
    gemm_xwT_bias<__bf16><<<gemm_blocks, 256, 0, stream>>>(qb, wvb, Wv_b, Vb, M, E_, E_);

    // 3) scores = Q K^T / 8 -> attn region of d_out. 64 blocks per head.
    attn_scores<<<dim3(64, B_ * H_), 256, 0, stream>>>(Qb, Kb, attn_f);

    // 4) softmax in place
    softmax_rows<<<(B_ * H_ * S_) / 8, 256, 0, stream>>>(attn_f);

    // 5) per-head transpose of V
    transpose_v<<<(int)(BSE / 256), 256, 0, stream>>>(Vb, Vt);

    // 6) ctx = attn @ V : 8 blocks (of 128 rows) per head
    attn_ctx<<<dim3(8, B_ * H_), 256, 0, stream>>>(attn_f, Vt, ctx);

    // 7) output projection -> fp32 d_out
    gemm_xwT_bias<float><<<gemm_blocks, 256, 0, stream>>>(ctx, wob, Wo_b, out_f, M, E_, E_);

    (void)in_sizes; (void)n_in; (void)out_size; (void)ws_size;
}
